// MultiHeadAttention_31009663877395
// MI455X (gfx1250) — matrix operand order repaired and checked by reading, not compiled, not run on hardware
//
#include <hip/hip_runtime.h>
#include <hip/hip_bf16.h>

// ---------------- problem constants ----------------
constexpr int kN   = 3072;   // nodes
constexpr int kHID = 512;    // hidden
constexpr int kH   = 8;      // heads
constexpr int kD   = 64;     // head dim
constexpr int kMT  = kN / 16;      // 192 row tiles
constexpr int kNTQ = kHID / 16;    // 32 col tiles for QKV gemm

typedef __attribute__((ext_vector_type(16))) _Float16 v16h;
typedef __attribute__((ext_vector_type(8)))  _Float16 h8;
typedef __attribute__((ext_vector_type(8)))  float    v8f;
typedef __attribute__((ext_vector_type(4)))  unsigned int u32x4;
typedef __attribute__((ext_vector_type(4)))  int      i32x4;
typedef __attribute__((ext_vector_type(8)))  int      i32x8;

#if __has_builtin(__builtin_amdgcn_tensor_load_to_lds) && \
    __has_builtin(__builtin_amdgcn_s_wait_tensorcnt)
#define USE_TDM 1
#else
#define USE_TDM 0
#endif

// ---------------- workspace layout (bytes) ----------------
constexpr size_t OFF_XH   = 0;                                  // x f16      [3072][512]
constexpr size_t OFF_WQH  = OFF_XH  + (size_t)kN*kHID*2;        // Wq f16     [512][512]
constexpr size_t OFF_WKH  = OFF_WQH + (size_t)kHID*kHID*2;      // Wk f16
constexpr size_t OFF_WVH  = OFF_WKH + (size_t)kHID*kHID*2;      // Wv f16
constexpr size_t OFF_WOH  = OFF_WVH + (size_t)kHID*kHID*2;      // Wo f16     [64][512]
constexpr size_t OFF_QH   = OFF_WOH + (size_t)kD*kHID*2;        // Q  f16     [H][N][D] (pre-scaled 1/8)
constexpr size_t OFF_KH   = OFF_QH  + (size_t)kH*kN*kD*2;       // K  f16     [H][N][D]
constexpr size_t OFF_VT   = OFF_KH  + (size_t)kH*kN*kD*2;       // V  f16     [H][D][N] (transposed)
constexpr size_t OFF_MI   = OFF_VT  + (size_t)kH*kN*kD*2;       // row max    [H][N] f32
constexpr size_t OFF_LI   = OFF_MI  + (size_t)kH*kN*4;          // row sumexp [H][N] f32
constexpr size_t OFF_ATT  = OFF_LI  + (size_t)kH*kN*4;          // attended f16 [N][512]
// total ~17.6 MB

// ---------------- helpers ----------------
__device__ inline v8f wmma_f16(v16h a, v16h b, v8f c) {
  return __builtin_amdgcn_wmma_f32_16x16x32_f16(false, a, false, b, (short)0, c,
                                                false, false);
}

// A-fragment (16x32 f16): lane row = lane%16; halves 0..7  -> k0 + g*16 + j
//                                    halves 8..15 -> k0 + g*16 + 8 + j
__device__ inline v16h load_a_frag(const _Float16* __restrict__ row, int k0, int g) {
  h8 lo = *(const h8*)(row + k0 + g * 16);
  h8 hi = *(const h8*)(row + k0 + g * 16 + 8);
  return __builtin_shufflevector(lo, hi, 0, 1, 2, 3, 4, 5, 6, 7,
                                         8, 9, 10, 11, 12, 13, 14, 15);
}

// Gaussian RBF spatial bias: relu(gw0*exp(..) + gw1*exp(..) + gb)
__device__ inline float gbias(float2 a, float2 b,
                              float mu0, float mu1, float iv0, float iv1,
                              float gw0, float gw1, float gb) {
  float dx = a.x - b.x, dy = a.y - b.y;
  float dist = sqrtf(dx * dx + dy * dy);
  float t0 = dist - mu0, t1 = dist - mu1;
  float v = gw0 * __expf(-0.5f * t0 * t0 * iv0) +
            gw1 * __expf(-0.5f * t1 * t1 * iv1) + gb;
  return v > 0.f ? v : 0.f;
}

#if USE_TDM
// TDM: DMA a 16-row x 64-half f16 tile (row stride = 64 halves) into LDS.
// D# built per CDNA5 ISA:
//   group0: count=1 | lds_addr | global_addr[56:0] | type=2
//   group1: data_size=1(2B), tensor_dim0=64, tensor_dim1=16, tile_dim0=64,
//           tile_dim1=16, tensor_dim0_stride=64
//   groups 2/3: zero (2-D tensor)
__device__ inline void tdm_load_k_tile(const _Float16* g, unsigned lds_off) {
  unsigned long long ga = (unsigned long long)(uintptr_t)g;
  u32x4 g0 = {0u, 0u, 0u, 0u};
  g0[0] = 1u;                                          // count = 1 (valid)
  g0[1] = lds_off;                                     // LDS byte address
  g0[2] = (unsigned)ga;                                // global_addr[31:0]
  g0[3] = ((unsigned)(ga >> 32) & 0x1FFFFFFu) | (2u << 30);  // addr[56:32]|type=2
  i32x8 g1 = {0, 0, 0, 0, 0, 0, 0, 0};
  g1[0] = 0x00010000;                                  // data_size = 1 -> 2 bytes
  g1[1] = (int)(64u << 16);                            // tensor_dim0 = 64
  g1[2] = (int)(16u << 16);                            // tensor_dim1 = 16
  g1[3] = (int)(64u << 16);                            // tile_dim0 = 64
  g1[4] = 16;                                          // tile_dim1 = 16
  g1[5] = 64;                                          // tensor_dim0_stride = 64
  i32x4 z4 = {0, 0, 0, 0};
#if defined(__clang_major__) && __clang_major__ >= 23
  i32x8 z8 = {0, 0, 0, 0, 0, 0, 0, 0};
  __builtin_amdgcn_tensor_load_to_lds(g0, g1, z4, z4, z8, 0);
#else
  __builtin_amdgcn_tensor_load_to_lds(g0, g1, z4, z4, 0);
#endif
}
#endif

// ---------------- kernel 0: f32 -> f16 convert ----------------
__global__ void cvt_f16(const float* __restrict__ src, _Float16* __restrict__ dst, int n) {
  int i = blockIdx.x * blockDim.x + threadIdx.x;
  if (i < n) dst[i] = (_Float16)src[i];
}

// ---------------- kernel 1: QKV projections (WMMA GEMM) ----------------
__global__ void qkv_gemm(const _Float16* __restrict__ xh,
                         const _Float16* __restrict__ wqh,
                         const _Float16* __restrict__ wkh,
                         const _Float16* __restrict__ wvh,
                         const float* __restrict__ qb,
                         const float* __restrict__ kb,
                         const float* __restrict__ vb,
                         _Float16* __restrict__ Qh,
                         _Float16* __restrict__ Kh,
                         _Float16* __restrict__ Vt) {
  int tid = threadIdx.x, wv = tid >> 5, lane = tid & 31, g = lane >> 4, c = lane & 15;
  int t = blockIdx.x * 8 + wv;
  int mat = t / (kMT * kNTQ);
  int r2  = t % (kMT * kNTQ);
  int m0 = (r2 >> 5) * 16, n0 = (r2 & 31) * 16;

  const _Float16* wh  = (mat == 0) ? wqh : (mat == 1) ? wkh : wvh;
  const float*    bpt = (mat == 0) ? qb  : (mat == 1) ? kb  : vb;

  const _Float16* arow = xh + (size_t)(m0 + c) * kHID;
  const _Float16* brow = wh + (size_t)(n0 + c) * kHID;

  v8f acc = {};
#pragma unroll 4
  for (int kt = 0; kt < 16; ++kt) {
    int k0 = kt * 32;
    v16h a = load_a_frag(arow, k0, g);
    v16h b = *(const v16h*)(brow + k0 + g * 16);
    acc = wmma_f16(a, b, acc);
  }
  float bb = bpt[n0 + c];
  int hh = (n0 + c) >> 6, dd = (n0 + c) & 63;
#pragma unroll
  for (int r = 0; r < 8; ++r) {
    int m = m0 + g * 8 + r;
    float v = acc[r] + bb;
    if (mat == 0)      Qh[((size_t)(hh * kN + m)) * kD + dd] = (_Float16)(v * 0.125f);
    else if (mat == 1) Kh[((size_t)(hh * kN + m)) * kD + dd] = (_Float16)v;
    else               Vt[((size_t)(hh * kD + dd)) * kN + m] = (_Float16)v;
  }
}

// ---------------- kernel 2: pass 1 — online softmax stats ----------------
// block = (head, 16-row query tile); 8 waves split the 192 key tiles.
// K tiles are staged LDS-side by the Tensor Data Mover, double buffered:
// DMA tile t+8 while WMMA+softmax consumes tile t.
__global__ void attn_pass1(const _Float16* __restrict__ Qh,
                           const _Float16* __restrict__ Kh,
                           const float2* __restrict__ loc,
                           const float* __restrict__ mu,
                           const float* __restrict__ sigma,
                           const float* __restrict__ gbf_w,
                           const float* __restrict__ gbf_b,
                           float* __restrict__ m_i,
                           float* __restrict__ l_i) {
  __shared__ float lmS[8][16], lsS[8][16];
#if USE_TDM
  __shared__ __align__(128) _Float16 ktile[8][2][16 * 64];  // 32 KB
#endif
  int tid = threadIdx.x, wv = tid >> 5, lane = tid & 31, g = lane >> 4, c = lane & 15;
  int h = blockIdx.x / kMT, mtile = blockIdx.x % kMT, m0 = mtile * 16;

  float mu0 = mu[0], mu1 = mu[1];
  float s0 = sigma[0], s1 = sigma[1];
  float iv0 = 1.f / (s0 * s0 + 1e-6f), iv1 = 1.f / (s1 * s1 + 1e-6f);
  float gw0 = gbf_w[h * 2], gw1 = gbf_w[h * 2 + 1], gb = gbf_b[h];

  const _Float16* qrow = Qh + ((size_t)h * kN + m0 + c) * kD;
  v16h aq0 = load_a_frag(qrow, 0, g);
  v16h aq1 = load_a_frag(qrow, 32, g);

  float2 ql[8];
#pragma unroll
  for (int r = 0; r < 8; ++r) ql[r] = loc[m0 + g * 8 + r];

  float lm[8], ls[8];
#pragma unroll
  for (int r = 0; r < 8; ++r) { lm[r] = -3.0e38f; ls[r] = 0.f; }

#if USE_TDM
  unsigned lbase[2];
  lbase[0] = (unsigned)(uintptr_t)&ktile[wv][0][0];
  lbase[1] = (unsigned)(uintptr_t)&ktile[wv][1][0];
  // prologue: DMA first tile into buffer 0
  tdm_load_k_tile(Kh + ((size_t)h * kN + wv * 16) * kD, lbase[0]);
  int buf = 0;
#endif

  for (int nt = wv; nt < kMT; nt += 8) {
    v16h b0, b1;
#if USE_TDM
    int ntn = nt + 8;
    if (ntn < kMT) {
      tdm_load_k_tile(Kh + ((size_t)h * kN + ntn * 16) * kD, lbase[buf ^ 1]);
      __builtin_amdgcn_s_wait_tensorcnt(1);   // oldest DMA (current tile) done
    } else {
      __builtin_amdgcn_s_wait_tensorcnt(0);
    }
    asm volatile("" ::: "memory");
    const _Float16* kt = &ktile[wv][buf][0];
    b0 = *(const v16h*)(kt + c * 64 + g * 16);
    b1 = *(const v16h*)(kt + c * 64 + 32 + g * 16);
    buf ^= 1;
#else
    const _Float16* krow = Kh + ((size_t)h * kN + nt * 16 + c) * kD;
    b0 = *(const v16h*)(krow + g * 16);
    b1 = *(const v16h*)(krow + 32 + g * 16);
#endif
    v8f acc = {};
    acc = wmma_f16(aq0, b0, acc);
    acc = wmma_f16(aq1, b1, acc);
    float2 kl = loc[nt * 16 + c];
#pragma unroll
    for (int r = 0; r < 8; ++r) {
      float s = acc[r] + gbias(ql[r], kl, mu0, mu1, iv0, iv1, gw0, gw1, gb);
      float mn = fmaxf(lm[r], s);
      ls[r] = ls[r] * __expf(lm[r] - mn) + __expf(s - mn);
      lm[r] = mn;
    }
  }
#if USE_TDM
  __builtin_amdgcn_s_wait_tensorcnt(0);       // drain the pipe
#endif
  // reduce (max,sum) across the 16 lanes that share each row
#pragma unroll
  for (int r = 0; r < 8; ++r) {
    for (int off = 1; off < 16; off <<= 1) {
      float om = __shfl_xor(lm[r], off, 32);
      float os = __shfl_xor(ls[r], off, 32);
      float mn = fmaxf(lm[r], om);
      ls[r] = ls[r] * __expf(lm[r] - mn) + os * __expf(om - mn);
      lm[r] = mn;
    }
  }
  if (c == 0) {
#pragma unroll
    for (int r = 0; r < 8; ++r) { lmS[wv][g * 8 + r] = lm[r]; lsS[wv][g * 8 + r] = ls[r]; }
  }
  __syncthreads();
  if (tid < 16) {
    float m = -3.0e38f, s = 0.f;
    for (int w2 = 0; w2 < 8; ++w2) {
      float om = lmS[w2][tid], os = lsS[w2][tid];
      float mn = fmaxf(m, om);
      s = s * __expf(m - mn) + os * __expf(om - mn);
      m = mn;
    }
    m_i[(size_t)h * kN + m0 + tid] = m;
    l_i[(size_t)h * kN + m0 + tid] = s;
  }
}

// ---------------- kernel 3: pass 2 — weights write + P·V ----------------
__global__ void attn_pass2(const _Float16* __restrict__ Qh,
                           const _Float16* __restrict__ Kh,
                           const _Float16* __restrict__ Vt,
                           const float2* __restrict__ loc,
                           const float* __restrict__ mu,
                           const float* __restrict__ sigma,
                           const float* __restrict__ gbf_w,
                           const float* __restrict__ gbf_b,
                           const float* __restrict__ m_i,
                           const float* __restrict__ l_i,
                           float* __restrict__ wout,          // [H][N][N]
                           _Float16* __restrict__ attended) { // [N][512]
  __shared__ _Float16 pbuf[8][16][32];
  __shared__ float obuf[8][16][64];
  int tid = threadIdx.x, wv = tid >> 5, lane = tid & 31, g = lane >> 4, c = lane & 15;
  int h = blockIdx.x / kMT, mtile = blockIdx.x % kMT, m0 = mtile * 16;

  float mu0 = mu[0], mu1 = mu[1];
  float s0 = sigma[0], s1 = sigma[1];
  float iv0 = 1.f / (s0 * s0 + 1e-6f), iv1 = 1.f / (s1 * s1 + 1e-6f);
  float gw0 = gbf_w[h * 2], gw1 = gbf_w[h * 2 + 1], gb = gbf_b[h];

  const _Float16* qrow = Qh + ((size_t)h * kN + m0 + c) * kD;
  v16h aq0 = load_a_frag(qrow, 0, g);
  v16h aq1 = load_a_frag(qrow, 32, g);

  float2 ql[8];
  float mrow[8], invl[8];
#pragma unroll
  for (int r = 0; r < 8; ++r) {
    int qm = m0 + g * 8 + r;
    ql[r]   = loc[qm];
    mrow[r] = m_i[(size_t)h * kN + qm];
    invl[r] = 1.f / l_i[(size_t)h * kN + qm];
  }

  v8f ov0 = {}, ov1 = {}, ov2 = {}, ov3 = {};
  float* wbase = wout + (size_t)h * kN * kN;

  for (int kc = wv; kc < kN / 32; kc += 8) {
    int key0 = kc * 32;
    if (key0 + 256 < kN)   // prefetch next chunk's K rows into near caches
      __builtin_prefetch(Kh + ((size_t)h * kN + key0 + 256) * kD, 0, 0);
#pragma unroll
    for (int t = 0; t < 2; ++t) {
      int kt0 = key0 + t * 16;
      const _Float16* krow = Kh + ((size_t)h * kN + kt0 + c) * kD;
      v16h b0 = *(const v16h*)(krow + g * 16);
      v16h b1 = *(const v16h*)(krow + 32 + g * 16);
      v8f acc = {};
      acc = wmma_f16(aq0, b0, acc);
      acc = wmma_f16(aq1, b1, acc);
      float2 kl = loc[kt0 + c];
#pragma unroll
      for (int r = 0; r < 8; ++r) {
        int qm = m0 + g * 8 + r;
        float s = acc[r] + gbias(ql[r], kl, mu0, mu1, iv0, iv1, gw0, gw1, gb);
        float p = __expf(s - mrow[r]) * invl[r];
        wbase[(size_t)qm * kN + kt0 + c] = p;          // final attention weight
        pbuf[wv][g * 8 + r][t * 16 + c]  = (_Float16)p;
      }
    }
    // wave-local LDS transpose C-layout -> A-layout; wait for our DS stores
    asm volatile("s_wait_dscnt 0" ::: "memory");
    h8 lo = *(const h8*)&pbuf[wv][c][g * 16];
    h8 hi = *(const h8*)&pbuf[wv][c][g * 16 + 8];
    v16h pa = __builtin_shufflevector(lo, hi, 0, 1, 2, 3, 4, 5, 6, 7,
                                              8, 9, 10, 11, 12, 13, 14, 15);
    const _Float16* vrow = Vt + ((size_t)(h * kD + c)) * kN + key0 + g * 16;
    ov0 = wmma_f16(pa, *(const v16h*)(vrow),                    ov0);
    ov1 = wmma_f16(pa, *(const v16h*)(vrow + 16 * (size_t)kN), ov1);
    ov2 = wmma_f16(pa, *(const v16h*)(vrow + 32 * (size_t)kN), ov2);
    ov3 = wmma_f16(pa, *(const v16h*)(vrow + 48 * (size_t)kN), ov3);
  }

  // combine the 8 waves' partial O tiles
#pragma unroll
  for (int r = 0; r < 8; ++r) {
    int row = g * 8 + r;
    obuf[wv][row][c]      = ov0[r];
    obuf[wv][row][16 + c] = ov1[r];
    obuf[wv][row][32 + c] = ov2[r];
    obuf[wv][row][48 + c] = ov3[r];
  }
  __syncthreads();
  for (int i = tid; i < 16 * 64; i += 256) {
    int row = i >> 6, col = i & 63;
    float s = 0.f;
#pragma unroll
    for (int w2 = 0; w2 < 8; ++w2) s += obuf[w2][row][col];
    attended[(size_t)(m0 + row) * kHID + h * kD + col] = (_Float16)s;
  }
}

// ---------------- kernel 4: output projection (WMMA GEMM) ----------------
__global__ void out_proj(const _Float16* __restrict__ attended,
                         const _Float16* __restrict__ woh,
                         const float* __restrict__ wob,
                         float* __restrict__ out) {
  int tid = threadIdx.x, wv = tid >> 5, lane = tid & 31, g = lane >> 4, c = lane & 15;
  int t = blockIdx.x * 8 + wv;
  int m0 = (t >> 2) * 16, n0 = (t & 3) * 16;
  const _Float16* arow = attended + (size_t)(m0 + c) * kHID;
  const _Float16* brow = woh + (size_t)(n0 + c) * kHID;
  v8f acc = {};
#pragma unroll 4
  for (int kt = 0; kt < 16; ++kt) {
    int k0 = kt * 32;
    v16h a = load_a_frag(arow, k0, g);
    v16h b = *(const v16h*)(brow + k0 + g * 16);
    acc = wmma_f16(a, b, acc);
  }
  float bb = wob[n0 + c];
#pragma unroll
  for (int r = 0; r < 8; ++r)
    out[(size_t)(m0 + g * 8 + r) * kD + n0 + c] = acc[r] + bb;
}

// ---------------- launcher ----------------
extern "C" void kernel_launch(void* const* d_in, const int* in_sizes, int n_in,
                              void* d_out, int out_size, void* d_ws, size_t ws_size,
                              hipStream_t stream) {
  const float* x     = (const float*)d_in[0];
  const float* loc   = (const float*)d_in[1];
  const float* mu    = (const float*)d_in[2];
  const float* sigma = (const float*)d_in[3];
  const float* gbf_w = (const float*)d_in[4];
  const float* gbf_b = (const float*)d_in[5];
  const float* Wq_w  = (const float*)d_in[6];
  const float* Wq_b  = (const float*)d_in[7];
  const float* Wk_w  = (const float*)d_in[8];
  const float* Wk_b  = (const float*)d_in[9];
  const float* Wv_w  = (const float*)d_in[10];
  const float* Wv_b  = (const float*)d_in[11];
  const float* Wo_w  = (const float*)d_in[12];
  const float* Wo_b  = (const float*)d_in[13];

  float* out = (float*)d_out;                      // [N*D] output, then [H*N*N] weights
  float* weights = out + (size_t)kN * kD;

  char* ws = (char*)d_ws;
  _Float16* xh  = (_Float16*)(ws + OFF_XH);
  _Float16* wqh = (_Float16*)(ws + OFF_WQH);
  _Float16* wkh = (_Float16*)(ws + OFF_WKH);
  _Float16* wvh = (_Float16*)(ws + OFF_WVH);
  _Float16* woh = (_Float16*)(ws + OFF_WOH);
  _Float16* Qh  = (_Float16*)(ws + OFF_QH);
  _Float16* Kh  = (_Float16*)(ws + OFF_KH);
  _Float16* Vt  = (_Float16*)(ws + OFF_VT);
  float*    m_i = (float*)(ws + OFF_MI);
  float*    l_i = (float*)(ws + OFF_LI);
  _Float16* att = (_Float16*)(ws + OFF_ATT);

  // 0) f32 -> f16 conversions
  cvt_f16<<<(kN * kHID + 255) / 256, 256, 0, stream>>>(x, xh, kN * kHID);
  cvt_f16<<<(kHID * kHID + 255) / 256, 256, 0, stream>>>(Wq_w, wqh, kHID * kHID);
  cvt_f16<<<(kHID * kHID + 255) / 256, 256, 0, stream>>>(Wk_w, wkh, kHID * kHID);
  cvt_f16<<<(kHID * kHID + 255) / 256, 256, 0, stream>>>(Wv_w, wvh, kHID * kHID);
  cvt_f16<<<(kD * kHID + 255) / 256, 256, 0, stream>>>(Wo_w, woh, kD * kHID);

  // 1) QKV projections: 3 * 192 * 32 tiles / 8 waves per block
  qkv_gemm<<<(3 * kMT * kNTQ) / 8, 256, 0, stream>>>(
      xh, wqh, wkh, wvh, Wq_b, Wk_b, Wv_b, Qh, Kh, Vt);

  // 2) pass 1: per-row softmax stats (online, flash-style, TDM double-buffered)
  attn_pass1<<<kH * kMT, 256, 0, stream>>>(
      Qh, Kh, (const float2*)loc, mu, sigma, gbf_w, gbf_b, m_i, l_i);

  // 3) pass 2: recompute scores, write normalized weights once, accumulate P·V
  attn_pass2<<<kH * kMT, 256, 0, stream>>>(
      Qh, Kh, Vt, (const float2*)loc, mu, sigma, gbf_w, gbf_b,
      m_i, l_i, weights, att);

  // 4) output projection
  out_proj<<<(kMT * 4) / 8, 256, 0, stream>>>(att, woh, Wo_b, out);
}
